// TopKSAE_47966194762191
// MI455X (gfx1250) — matrix operand order repaired and checked by reading, not compiled, not run on hardware
//
#include <hip/hip_runtime.h>

// ---------- problem sizes ----------
#define B_ROWS 8192
#define D_DIM  768
#define H_DIM  24576
#define K_SEL  64

// ---------- GEMM tiling ----------
#define BM 128
#define BN 128
#define BK 32
#define NK (D_DIM / BK)      // 24 k-steps
#define LDS_K 40             // bf16 elems per LDS row (32 + 8 pad = 80B, 16B aligned)

typedef __attribute__((ext_vector_type(16))) __bf16 bf16x16;
typedef __attribute__((ext_vector_type(8)))  __bf16 bf16x8;
typedef __attribute__((ext_vector_type(8)))  float  v8f;
typedef __attribute__((ext_vector_type(4)))  float  v4f;
typedef __attribute__((ext_vector_type(8)))  unsigned short us8;
typedef int v4i __attribute__((vector_size(16)));   // matches builtin param type

// Detect gfx1250 async global->LDS builtins (ASYNCcnt path)
#if defined(__has_builtin)
#if __has_builtin(__builtin_amdgcn_global_load_async_to_lds_b128) && \
    __has_builtin(__builtin_amdgcn_s_wait_asynccnt)
#define SAE_ASYNC 1
#endif
#endif
#ifndef SAE_ASYNC
#define SAE_ASYNC 0
#endif

#if SAE_ASYNC
// imm offset is added to BOTH global and LDS addresses (ISA 08_async_tensor §4.4)
#define ASYNC_CP16(gp, lp, off)                                          \
    __builtin_amdgcn_global_load_async_to_lds_b128(                      \
        (__attribute__((address_space(1))) v4i*)(gp),                    \
        (__attribute__((address_space(3))) v4i*)(lp), (off), 0)
#endif

// fp32 -> bf16 round-to-nearest-even, pure integer
__device__ __forceinline__ unsigned short f2bf(float x) {
    unsigned u = __float_as_uint(x);
    unsigned r = u + 0x7FFFu + ((u >> 16) & 1u);
    return (unsigned short)(r >> 16);
}

// A-fragment: lane<16 needs K {0..7, 16..23}; lane>=16 needs K {8..15, 24..31}
__device__ __forceinline__ bf16x16 load_frag_a(const unsigned short* rowp, int q) {
    bf16x8 lo = *(const bf16x8*)(rowp + q * 8);
    bf16x8 hi = *(const bf16x8*)(rowp + (q + 2) * 8);
    return __builtin_shufflevector(lo, hi, 0,1,2,3,4,5,6,7,8,9,10,11,12,13,14,15);
}

// B-fragment: lane<16 needs K {0..7, 16..23}; lane>=16 needs K {8..15, 24..31}
__device__ __forceinline__ bf16x16 load_frag_b(const unsigned short* rowp, int q) {
    bf16x8 lo = *(const bf16x8*)(rowp + q * 8);
    bf16x8 hi = *(const bf16x8*)(rowp + (q + 2) * 8);
    return __builtin_shufflevector(lo, hi, 0,1,2,3,4,5,6,7,8,9,10,11,12,13,14,15);
}

// ==================== kernel 0: bulk fp32 -> bf16 ====================
__global__ void __launch_bounds__(256) sae_f32_to_bf16(const float* __restrict__ src,
                                                       unsigned short* __restrict__ dst,
                                                       int n) {
    int i = (blockIdx.x * 256 + threadIdx.x) * 8;
    if (i + 7 < n) {
        v4f a = *(const v4f*)(src + i);
        v4f b = *(const v4f*)(src + i + 4);
        us8 h;
#pragma unroll
        for (int e = 0; e < 4; ++e) { h[e] = f2bf(a[e]); h[e + 4] = f2bf(b[e]); }
        *(us8*)(dst + i) = h;
    }
}

// ==================== kernel 1: transpose dec_w [D,H] -> dec_wt [H,D] ====================
__global__ void __launch_bounds__(256) sae_transpose(const float* __restrict__ dw,
                                                     float* __restrict__ dwt) {
    __shared__ float t[32][33];
    const int tx = threadIdx.x & 31;
    const int ty = threadIdx.x >> 5;
    const int h0 = blockIdx.x * 32;
    const int d0 = blockIdx.y * 32;
#pragma unroll
    for (int yy = ty; yy < 32; yy += 8)
        t[yy][tx] = dw[(size_t)(d0 + yy) * H_DIM + h0 + tx];
    __syncthreads();
#pragma unroll
    for (int yy = ty; yy < 32; yy += 8)
        dwt[(size_t)(h0 + yy) * D_DIM + d0 + tx] = t[tx][yy];
}

// ==================== kernel 2: encode GEMM (bf16 WMMA, fp32 accum) ====================
// A: [B,D] bf16, W: [H,D] bf16, staging via async global->LDS when available.
__global__ void __launch_bounds__(256) sae_encode_gemm(const unsigned short* __restrict__ A,
                                                       const unsigned short* __restrict__ W,
                                                       const float* __restrict__ bias,
                                                       float* __restrict__ C) {
    __shared__ unsigned short sA[2][BM * LDS_K];
    __shared__ unsigned short sB[2][BN * LDS_K];

    const int tid  = threadIdx.x;
    const int lane = tid & 31;
    const int wav  = tid >> 5;
    const int wm   = wav >> 2;          // 0..1  (64 rows each)
    const int wn   = wav & 3;           // 0..3  (32 cols each)
    const int bM   = blockIdx.y * BM;
    const int bN   = blockIdx.x * BN;

    // staging map: 2 threads per tile row, 32 bytes (16 bf16) each
    const int lrow  = tid >> 1;         // 0..127
    const int lhalf = (tid & 1) * 16;   // bf16 element offset within row
    const unsigned short* gA = A + (size_t)(bM + lrow) * D_DIM + lhalf;
    const unsigned short* gB = W + (size_t)(bN + lrow) * D_DIM + lhalf;
    const int lofs = lrow * LDS_K + lhalf;

    v8f acc[4][2];
    const v8f vzero = {0.f, 0.f, 0.f, 0.f, 0.f, 0.f, 0.f, 0.f};
#pragma unroll
    for (int mt = 0; mt < 4; ++mt)
#pragma unroll
        for (int nt = 0; nt < 2; ++nt) acc[mt][nt] = vzero;

    // ---- prologue: stage k-step 0 into buffer 0 ----
#if SAE_ASYNC
    ASYNC_CP16(gA, &sA[0][lofs], 0);
    ASYNC_CP16(gA, &sA[0][lofs], 16);
    ASYNC_CP16(gB, &sB[0][lofs], 0);
    ASYNC_CP16(gB, &sB[0][lofs], 16);
    __builtin_amdgcn_s_wait_asynccnt(0);
#else
    {
        us8 a0 = *(const us8*)(gA);
        us8 a1 = *(const us8*)(gA + 8);
        us8 b0 = *(const us8*)(gB);
        us8 b1 = *(const us8*)(gB + 8);
        *(us8*)(&sA[0][lofs])     = a0;
        *(us8*)(&sA[0][lofs + 8]) = a1;
        *(us8*)(&sB[0][lofs])     = b0;
        *(us8*)(&sB[0][lofs + 8]) = b1;
    }
#endif
    __syncthreads();

    const int q  = lane >> 4;
    const int lm = lane & 15;

    for (int kt = 0; kt < NK; ++kt) {
        const int cur = kt & 1;
        const bool have_next = (kt + 1 < NK);

#if SAE_ASYNC
        if (have_next) {
            const int nxt = 1 - cur;
            const unsigned short* pA = gA + (kt + 1) * BK;
            const unsigned short* pB = gB + (kt + 1) * BK;
            ASYNC_CP16(pA, &sA[nxt][lofs], 0);
            ASYNC_CP16(pA, &sA[nxt][lofs], 16);
            ASYNC_CP16(pB, &sB[nxt][lofs], 0);
            ASYNC_CP16(pB, &sB[nxt][lofs], 16);
        }
#else
        us8 ra0, ra1, rb0, rb1;
        if (have_next) {
            const unsigned short* pA = gA + (kt + 1) * BK;
            const unsigned short* pB = gB + (kt + 1) * BK;
            ra0 = *(const us8*)(pA);
            ra1 = *(const us8*)(pA + 8);
            rb0 = *(const us8*)(pB);
            rb1 = *(const us8*)(pB + 8);
        }
#endif

        bf16x16 af[4], bfg[2];
#pragma unroll
        for (int mt = 0; mt < 4; ++mt)
            af[mt] = load_frag_a(&sA[cur][(wm * 64 + mt * 16 + lm) * LDS_K], q);
#pragma unroll
        for (int nt = 0; nt < 2; ++nt)
            bfg[nt] = load_frag_b(&sB[cur][(wn * 32 + nt * 16 + lm) * LDS_K], q);

#pragma unroll
        for (int mt = 0; mt < 4; ++mt)
#pragma unroll
            for (int nt = 0; nt < 2; ++nt)
                acc[mt][nt] = __builtin_amdgcn_wmma_f32_16x16x32_bf16(
                    false, af[mt], false, bfg[nt], (short)0, acc[mt][nt], false, false);

#if SAE_ASYNC
        __builtin_amdgcn_s_wait_asynccnt(0);
#else
        if (have_next) {
            const int nxt = 1 - cur;
            *(us8*)(&sA[nxt][lofs])     = ra0;
            *(us8*)(&sA[nxt][lofs + 8]) = ra1;
            *(us8*)(&sB[nxt][lofs])     = rb0;
            *(us8*)(&sB[nxt][lofs + 8]) = rb1;
        }
#endif
        __syncthreads();
    }

    // epilogue: VGPR r holds (M = r + 8*(lane>>4), N = lane&15)
    const int lg = lane >> 4;
#pragma unroll
    for (int mt = 0; mt < 4; ++mt)
#pragma unroll
        for (int nt = 0; nt < 2; ++nt) {
            const int gc  = bN + wn * 32 + nt * 16 + lm;
            const float bv = bias[gc];
            const int gr0 = bM + wm * 64 + mt * 16 + lg * 8;
            v8f a = acc[mt][nt];
#pragma unroll
            for (int r = 0; r < 8; ++r)
                C[(size_t)(gr0 + r) * H_DIM + gc] = a[r] + bv;
        }
}

// ==================== kernel 3: per-row top-K via radix select ====================
#define TOPK_CAP 4096

__device__ __forceinline__ unsigned mono_key(unsigned u) {
    return (u & 0x80000000u) ? ~u : (u | 0x80000000u);
}
__device__ __forceinline__ float key_to_float(unsigned k) {
    unsigned u = (k & 0x80000000u) ? (k & 0x7FFFFFFFu) : ~k;
    return __uint_as_float(u);
}

__global__ void __launch_bounds__(256) sae_topk(const float* __restrict__ enc,
                                                int* __restrict__ oidx,
                                                float* __restrict__ oval) {
    const int row = blockIdx.x;
    const int tid = threadIdx.x;
    __shared__ unsigned hist[256];
    __shared__ unsigned ckey[TOPK_CAP];
    __shared__ int      cidx[TOPK_CAP];
    __shared__ unsigned s_cand, s_out, s_tie, s_prefix, s_need;

    const float* rowp = enc + (size_t)row * H_DIM;
    int* ri = oidx + (size_t)row * K_SEL;
    float* rv = oval + (size_t)row * K_SEL;

    hist[tid] = 0;
    if (tid == 0) { s_cand = 0; s_out = 0; s_tie = 0; }
    __syncthreads();

    // pass 1: high-byte histogram
    for (int j = tid; j < H_DIM; j += 256) {
        unsigned key = mono_key(__float_as_uint(rowp[j]));
        atomicAdd(&hist[key >> 24], 1u);
    }
    __syncthreads();
    if (tid == 0) {
        unsigned cum = 0;
        for (int b = 255; b >= 0; --b) {
            unsigned c = hist[b];
            if (cum + c >= (unsigned)K_SEL) { s_prefix = (unsigned)b << 24; s_need = K_SEL - cum; break; }
            cum += c;
        }
    }
    __syncthreads();
    const unsigned bstar = s_prefix >> 24;

    // pass 2: emit definite members, collect boundary-bin candidates
    for (int j = tid; j < H_DIM; j += 256) {
        float x = rowp[j];
        unsigned key = mono_key(__float_as_uint(x));
        unsigned hb  = key >> 24;
        if (hb > bstar) {
            unsigned p = atomicAdd(&s_out, 1u);
            if (p < K_SEL) { ri[p] = j; rv[p] = x; }
        } else if (hb == bstar) {
            unsigned p = atomicAdd(&s_cand, 1u);
            if (p < TOPK_CAP) { ckey[p] = key; cidx[p] = j; }
        }
    }
    __syncthreads();
    const unsigned ncand = (s_cand < TOPK_CAP) ? s_cand : TOPK_CAP;

    // rounds on bytes 2,1,0 restricted to candidates
    for (int shift = 16; shift >= 0; shift -= 8) {
        hist[tid] = 0;
        __syncthreads();
        const unsigned pref  = s_prefix;
        const unsigned hmask = 0xFFFFFFFFu << (shift + 8);
        for (unsigned c = tid; c < ncand; c += 256) {
            unsigned key = ckey[c];
            if ((key & hmask) == pref) atomicAdd(&hist[(key >> shift) & 255u], 1u);
        }
        __syncthreads();
        if (tid == 0) {
            unsigned need = s_need, cum = 0;
            for (int b = 255; b >= 0; --b) {
                unsigned c = hist[b];
                if (cum + c >= need) { s_prefix = pref | ((unsigned)b << shift); s_need = need - cum; break; }
                cum += c;
            }
        }
        __syncthreads();
    }
    const unsigned T = s_prefix, ties = s_need;

    // final collect with exact tie handling
    for (unsigned c = tid; c < ncand; c += 256) {
        unsigned key = ckey[c];
        if (key > T) {
            unsigned p = atomicAdd(&s_out, 1u);
            if (p < K_SEL) { ri[p] = cidx[c]; rv[p] = key_to_float(key); }
        } else if (key == T) {
            if (atomicAdd(&s_tie, 1u) < ties) {
                unsigned p = atomicAdd(&s_out, 1u);
                if (p < K_SEL) { ri[p] = cidx[c]; rv[p] = key_to_float(key); }
            }
        }
    }
    __syncthreads();
    for (unsigned p = s_out + tid; p < (unsigned)K_SEL; p += 256) { ri[p] = 0; rv[p] = 0.f; }
}

// ==================== kernel 4: sparse decode ====================
__global__ void __launch_bounds__(256) sae_decode(const float* __restrict__ dwt,
                                                  const float* __restrict__ decb,
                                                  const int* __restrict__ oidx,
                                                  const float* __restrict__ oval,
                                                  float* __restrict__ recon) {
    const int row = blockIdx.x;
    const int tid = threadIdx.x;
    __shared__ int   sidx[K_SEL];
    __shared__ float sval[K_SEL];
    if (tid < K_SEL) {
        sidx[tid] = oidx[(size_t)row * K_SEL + tid];
        sval[tid] = oval[(size_t)row * K_SEL + tid];
    }
    __syncthreads();
    float a0 = decb[tid], a1 = decb[tid + 256], a2 = decb[tid + 512];
#pragma unroll 4
    for (int k = 0; k < K_SEL; ++k) {
        const float* wr = dwt + (size_t)sidx[k] * D_DIM;
        const float v = sval[k];
        a0 += v * wr[tid];
        a1 += v * wr[tid + 256];
        a2 += v * wr[tid + 512];
    }
    float* out = recon + (size_t)row * D_DIM;
    out[tid] = a0; out[tid + 256] = a1; out[tid + 512] = a2;
}

// ==================== launcher ====================
extern "C" void kernel_launch(void* const* d_in, const int* in_sizes, int n_in,
                              void* d_out, int out_size, void* d_ws, size_t ws_size,
                              hipStream_t stream) {
    const float* emb  = (const float*)d_in[0];   // [B, D]
    const float* encw = (const float*)d_in[1];   // [H, D]
    const float* encb = (const float*)d_in[2];   // [H]
    const float* decw = (const float*)d_in[3];   // [D, H]
    const float* decb = (const float*)d_in[4];   // [D]

    float* enc   = (float*)d_out;                               // [B, H]
    float* recon = (float*)d_out + (size_t)B_ROWS * H_DIM;      // [B, D]

    char* ws = (char*)d_ws;
    size_t off = 0;
    float* dwt   = (float*)(ws + off);  off += (size_t)H_DIM * D_DIM * sizeof(float);     // 75.5 MB
    int*   tidx  = (int*)(ws + off);    off += (size_t)B_ROWS * K_SEL * sizeof(int);      // 2 MB
    float* tval  = (float*)(ws + off);  off += (size_t)B_ROWS * K_SEL * sizeof(float);    // 2 MB
    unsigned short* embh  = (unsigned short*)(ws + off); off += (size_t)B_ROWS * D_DIM * 2; // 12.6 MB
    unsigned short* encwh = (unsigned short*)(ws + off); off += (size_t)H_DIM  * D_DIM * 2; // 37.7 MB

    const int n_emb  = B_ROWS * D_DIM;
    const int n_encw = H_DIM * D_DIM;
    sae_f32_to_bf16<<<dim3(n_emb  / (256 * 8)), 256, 0, stream>>>(emb,  embh,  n_emb);
    sae_f32_to_bf16<<<dim3(n_encw / (256 * 8)), 256, 0, stream>>>(encw, encwh, n_encw);
    sae_transpose<<<dim3(H_DIM / 32, D_DIM / 32), 256, 0, stream>>>(decw, dwt);
    sae_encode_gemm<<<dim3(H_DIM / BN, B_ROWS / BM), 256, 0, stream>>>(embh, encwh, encb, enc);
    sae_topk<<<dim3(B_ROWS), 256, 0, stream>>>(enc, tidx, tval);
    sae_decode<<<dim3(B_ROWS), 256, 0, stream>>>(dwt, decb, tidx, tval, recon);
}
